// PointVoxelCNN_70050916598405
// MI455X (gfx1250) — matrix operand order repaired and checked by reading, not compiled, not run on hardware
//
#include <hip/hip_runtime.h>

typedef _Float16 f16;
typedef __attribute__((ext_vector_type(16))) _Float16 v16h;
typedef __attribute__((ext_vector_type(8)))  _Float16 v8h;
typedef __attribute__((ext_vector_type(8)))  float    v8f;
typedef __attribute__((ext_vector_type(4)))  int      v4i;

#define B_ 8
#define N_ 32768
#define C_ 64
#define R_ 32
#define V_ (R_*R_*R_)          /* 32768 voxels */
#define INV_SQRT2 0.70710678118654752440f

#if __has_builtin(__builtin_amdgcn_global_load_async_to_lds_b128) && \
    __has_builtin(__builtin_amdgcn_s_wait_asynccnt)
#define PV_ASYNC_LDS 1
#else
#define PV_ASYNC_LDS 0
#endif

#define PV_AS1 __attribute__((address_space(1)))
#define PV_AS3 __attribute__((address_space(3)))

// ---------------------------------------------------------------------------
// WMMA fragment loaders (gfx1250 16x16x32 f16 layouts per CDNA5 ISA 7.12.2)
//  A (16x32, MxK): lane m = lane&15; lanes 0-15 hold K 0-7 (elems 0-7) and
//  K 16-23 (elems 8-15); lanes 16-31 hold K 8-15 / 24-31.
//  B (32x16, KxN): lane n = lane&15; same K order as A (lanes 0-15 K 0-7 /
//  16-23, lanes 16-31 K 8-15 / 24-31). Weights are pre-packed so each lane's 16 halfs
//  are contiguous (two b128 loads).
// ---------------------------------------------------------------------------
__device__ __forceinline__ v16h load_a_frag(const f16* __restrict__ row, int kc, int khalf) {
  const int k0 = kc * 32 + khalf * 8;
  v8h lo = *(const v8h*)(row + k0);
  v8h hi = *(const v8h*)(row + k0 + 16);
  v16h a;
#pragma unroll
  for (int e = 0; e < 8; ++e) { a[e] = lo[e]; a[e + 8] = hi[e]; }
  return a;
}

__device__ __forceinline__ v16h load_b_frag(const f16* __restrict__ p) {
  v8h lo = *(const v8h*)(p);
  v8h hi = *(const v8h*)(p + 8);
  v16h b;
#pragma unroll
  for (int e = 0; e < 8; ++e) { b[e] = lo[e]; b[e + 8] = hi[e]; }
  return b;
}

__device__ __forceinline__ void corners3(float px, float py, float pz,
                                         int& ix, int& iy, int& iz,
                                         float& fx, float& fy, float& fz) {
  float cx = fminf(fmaxf((px + 1.f) * 0.5f * 31.f, 0.f), 31.f);
  float cy = fminf(fmaxf((py + 1.f) * 0.5f * 31.f, 0.f), 31.f);
  float cz = fminf(fmaxf((pz + 1.f) * 0.5f * 31.f, 0.f), 31.f);
  float lx = fminf(floorf(cx), 30.f);
  float ly = fminf(floorf(cy), 30.f);
  float lz = fminf(floorf(cz), 30.f);
  fx = cx - lx; fy = cy - ly; fz = cz - lz;
  ix = (int)lx; iy = (int)ly; iz = (int)lz;
}

// ---------------------------------------------------------------------------
// Utility kernels
// ---------------------------------------------------------------------------
__global__ void pv_zero_kernel(float* __restrict__ p, long long n) {
  long long i = (long long)blockIdx.x * blockDim.x + threadIdx.x;
  long long stride = (long long)gridDim.x * blockDim.x;
  for (; i < n; i += stride) p[i] = 0.f;
}

// pack a 64(K) x 64(N) f32 matrix into WMMA-B fragment order
__global__ void pv_pack_mat64_kernel(const float* __restrict__ src, f16* __restrict__ dst) {
  int i = blockIdx.x * 256 + threadIdx.x;          // 4096 entries
  if (i >= 4096) return;
  int e = i & 15, lane = (i >> 4) & 31, nt = (i >> 9) & 3, kc = i >> 11;
  int k = kc * 32 + (lane >> 4) * 8 + (e & 7) + (e >> 3) * 16;
  int n = nt * 16 + (lane & 15);
  dst[i] = (f16)src[k * 64 + n];
}

// pack conv weights [O=64][I=64][27] into per-(tap,kc,ntile) B fragments
__global__ void pv_pack_convw_kernel(const float* __restrict__ src, f16* __restrict__ dst) {
  int i = blockIdx.x * 256 + threadIdx.x;          // 27*4096 = 110592 entries
  if (i >= 110592) return;
  int e = i & 15, lane = (i >> 4) & 31, nt = (i >> 9) & 3, kc = (i >> 11) & 1, tap = i >> 12;
  int cin = kc * 32 + (lane >> 4) * 8 + (e & 7) + (e >> 3) * 16;
  int n = nt * 16 + (lane & 15);
  dst[i] = (f16)src[(n * 64 + cin) * 27 + tap];
}

// ---------------------------------------------------------------------------
// Voxelize: trilinear scatter (one thread per point-corner)
// ---------------------------------------------------------------------------
__global__ __launch_bounds__(256)
void pv_scatter_kernel(const float* __restrict__ points, const float* __restrict__ features,
                       float* __restrict__ accum, float* __restrict__ cnt) {
  long long idx = (long long)blockIdx.x * 256 + threadIdx.x;   // B*N*8
  if (idx >= (long long)B_ * N_ * 8) return;
  int k = (int)(idx & 7);
  long long pi = idx >> 3;                                     // b*N + n
  int b = (int)(pi >> 15);
  const float* pp = points + pi * 3;
  int ix, iy, iz; float fx, fy, fz;
  corners3(pp[0], pp[1], pp[2], ix, iy, iz, fx, fy, fz);
  int ox = (k >> 2) & 1, oy = (k >> 1) & 1, oz = k & 1;
  float wk = (ox ? fx : 1.f - fx) * (oy ? fy : 1.f - fy) * (oz ? fz : 1.f - fz);
  long long v = ((long long)(ix + ox) * R_ + (iy + oy)) * R_ + (iz + oz);
  long long gidx = ((long long)b * V_ + v);
  float* ap = accum + gidx * 64;
  const float* fp = features + pi * 64;
#pragma unroll 8
  for (int c = 0; c < 64; ++c) atomicAdd(ap + c, wk * fp[c]);
  atomicAdd(cnt + gidx, wk);
}

// ---------------------------------------------------------------------------
// GroupNorm stats: optional cnt-normalization pass + per-(b,group) sum/sumsq
// Group g = channels {2g, 2g+1}; block = 64 voxels x 64 channels
// ---------------------------------------------------------------------------
__global__ __launch_bounds__(256)
void pv_gn_stats_kernel(const float* __restrict__ src, const float* __restrict__ cnt,
                        float* __restrict__ dstgrid, float* __restrict__ stats_acc) {
  __shared__ float sSum[256], sSq[256];
  __shared__ float cSum[64], cSq[64];
  int blk = blockIdx.x;                 // B * 512
  int b = blk >> 9;
  int vbase = (blk & 511) * 64;
  int tid = threadIdx.x;
  int c = tid & 63, vs = tid >> 6;
  long long base = (long long)b * V_ + vbase;
  float s = 0.f, q = 0.f;
  for (int v = vs; v < 64; v += 4) {
    long long vi = base + v;
    float val = src[vi * 64 + c];
    if (cnt) {
      val = val / fmaxf(cnt[vi], 1e-8f);
      dstgrid[vi * 64 + c] = val;
    }
    s += val; q += val * val;
  }
  sSum[tid] = s; sSq[tid] = q;
  __syncthreads();
  if (tid < 64) {
    cSum[tid] = sSum[tid] + sSum[tid + 64] + sSum[tid + 128] + sSum[tid + 192];
    cSq[tid]  = sSq[tid]  + sSq[tid + 64]  + sSq[tid + 128]  + sSq[tid + 192];
  }
  __syncthreads();
  if (tid < 32) {
    atomicAdd(&stats_acc[(b * 32 + tid) * 2 + 0], cSum[2 * tid] + cSum[2 * tid + 1]);
    atomicAdd(&stats_acc[(b * 32 + tid) * 2 + 1], cSq[2 * tid]  + cSq[2 * tid + 1]);
  }
}

__global__ void pv_gn_finalize_kernel(const float* __restrict__ acc, float* __restrict__ fin) {
  int i = threadIdx.x;                  // 256 = B*G
  if (i < 256) {
    const float n = 65536.f;            // (C/G) * R^3 = 2 * 32768
    float mean = acc[i * 2] / n;
    float var  = acc[i * 2 + 1] / n - mean * mean;
    fin[i * 2]     = mean;
    fin[i * 2 + 1] = rsqrtf(var + 1e-5f);
  }
}

// Apply GN + leaky, convert to f16 activations
__global__ __launch_bounds__(256)
void pv_act_kernel(const float* __restrict__ src, const float* __restrict__ fin,
                   const float* __restrict__ gamma, const float* __restrict__ beta,
                   f16* __restrict__ act) {
  long long i = (long long)blockIdx.x * 256 + threadIdx.x;   // B*V*64 = 16M
  if (i >= (long long)B_ * V_ * 64) return;
  int c = (int)(i & 63);
  long long bv = i >> 6;
  int b = (int)(bv >> 15);
  int g = c >> 1;
  float mean = fin[(b * 32 + g) * 2];
  float rstd = fin[(b * 32 + g) * 2 + 1];
  float v = (src[i] - mean) * rstd * gamma[c] + beta[c];
  v = v > 0.f ? v : 0.01f * v;
  act[i] = (f16)v;
}

// ---------------------------------------------------------------------------
// 3x3x3 SAME conv (64->64) as implicit GEMM with WMMA.
// Block: 128 threads = 4 waves, owns a full 32-voxel z-line (two 16-z tiles,
// double-buffered). Tile staging uses GLOBAL_LOAD_ASYNC_TO_LDS_B128 when
// available: tile1's async copies overlap tile0's 216 WMMAs (ASYNCcnt path).
// LDS: 2 x (3x3x18 halo x 64 ch f16) = 41.5 KB. K = 27 taps x 64 cin.
// ---------------------------------------------------------------------------
__global__ __launch_bounds__(128)
void pv_conv3d_wmma_kernel(const f16* __restrict__ act, const f16* __restrict__ wpack,
                           const float* __restrict__ bias, const float* __restrict__ resid,
                           float* __restrict__ out) {
  __shared__ __align__(16) f16 sAct[2][162 * 64];
  const int blk = blockIdx.x;              // B * 1024
  const int b = blk >> 10;
  const int r = blk & 1023;
  const int d = r >> 5;
  const int h = r & 31;
  const int tid = threadIdx.x;
  const long long gbase = (long long)b * ((long long)V_ * 64);

  const int lane = tid & 31;
  const int w = tid >> 5;                  // N-tile (16 out channels per wave)
  const int m_lo = lane & 15;
  const int khalf = lane >> 4;
  const int ch = w * 16 + m_lo;            // D: column N = lane&15
  const float bb = bias[ch];

  // stage halo tile zt (z0 = 16*zt): 162 rows x 128 bytes, 16B chunks
  auto stage = [&](int zt) {
    const int z0 = zt * 16;
    for (int idx = tid; idx < 162 * 8; idx += 128) {
      int row = idx >> 3, q = idx & 7;
      int dd = row / 54, rem = row % 54;
      int hh = rem / 18, zz = rem % 18;
      int di = d + dd - 1, hi2 = h + hh - 1, zi = z0 + zz - 1;
      f16* ldst = &sAct[zt][row * 64 + q * 8];
      if ((unsigned)di < 32u && (unsigned)hi2 < 32u && (unsigned)zi < 32u) {
        long long v = ((long long)di * R_ + hi2) * R_ + zi;
        const f16* gsrc = act + gbase + v * 64 + q * 8;
#if PV_ASYNC_LDS
        __builtin_amdgcn_global_load_async_to_lds_b128(
            (PV_AS1 v4i*)gsrc, (PV_AS3 v4i*)ldst, 0, 0);
#else
        *(uint4*)ldst = *(const uint4*)gsrc;
#endif
      } else {
        *(uint4*)ldst = make_uint4(0u, 0u, 0u, 0u);
      }
    }
  };

  // compute tile zt: 27 taps x 2 K-chunks = 54 WMMAs per wave
  auto compute = [&](int zt) {
    const int z0 = zt * 16;
    v8f acc = {};
    for (int tap = 0; tap < 27; ++tap) {
      int kd = tap / 9;
      int kh = (tap / 3) % 3;
      int kz = tap % 3;
      const f16* arow = &sAct[zt][(kd * 54 + kh * 18 + kz + m_lo) * 64];
#pragma unroll
      for (int kc = 0; kc < 2; ++kc) {
        v16h A = load_a_frag(arow, kc, khalf);
        v16h Bf = load_b_frag(wpack + ((((tap * 2 + kc) * 4 + w) * 32 + lane) << 4));
        acc = __builtin_amdgcn_wmma_f32_16x16x32_f16(false, A, false, Bf,
                                                     (short)0, acc, false, false);
      }
    }
#pragma unroll
    for (int rr = 0; rr < 8; ++rr) {
      int m = rr + khalf * 8;              // D: row M = r + (lane>>4)*8
      long long v = ((long long)d * R_ + h) * R_ + (z0 + m);
      long long oi = gbase + v * 64 + ch;
      float val = acc[rr] + bb;
      if (resid) val = (val + resid[oi]) * INV_SQRT2;
      out[oi] = val;
    }
  };

  stage(0);
#if PV_ASYNC_LDS
  __builtin_amdgcn_s_wait_asynccnt(0);
#endif
  __syncthreads();

  stage(1);        // async copies for tile1 fly while tile0's WMMAs run
  compute(0);

#if PV_ASYNC_LDS
  __builtin_amdgcn_s_wait_asynccnt(0);
#endif
  __syncthreads();

  compute(1);
}

// ---------------------------------------------------------------------------
// Fused point branch (3 WMMA GEMMs) + trilinear devoxelize + final combine.
// Block: 128 threads = 4 waves, 64-point tile x 64 channels.
// ---------------------------------------------------------------------------
__global__ __launch_bounds__(128)
void pv_point_devox_kernel(const float* __restrict__ points, const float* __restrict__ features,
                           const f16* __restrict__ wpi, const f16* __restrict__ wp1,
                           const f16* __restrict__ wp2,
                           const float* __restrict__ bpi, const float* __restrict__ bp1,
                           const float* __restrict__ bp2,
                           const float* __restrict__ out_grid, float* __restrict__ out) {
  __shared__ __align__(16) f16 sA[64 * 64];
  __shared__ __align__(16) f16 sB[64 * 64];
  __shared__ float sP[64 * 64];
  int blk = blockIdx.x;                 // B * 512
  int b = blk >> 9;
  int p0 = (blk & 511) * 64;
  int tid = threadIdx.x;
  int lane = tid & 31;
  int w = tid >> 5;
  int m_lo = lane & 15;
  int khalf = lane >> 4;
  int ch = w * 16 + m_lo;

  long long fbase = ((long long)b * N_ + p0) * 64;
  for (int idx = tid; idx < 4096; idx += 128)
    sA[idx] = (f16)features[fbase + idx];
  __syncthreads();

  float f0[4][8];

  // GEMM1: f0 = features @ w_pi + b_pi ; write leaky(f0) -> sB
  {
    float bb = bpi[ch];
#pragma unroll
    for (int mt = 0; mt < 4; ++mt) {
      v8f acc = {};
#pragma unroll
      for (int kc = 0; kc < 2; ++kc) {
        v16h A = load_a_frag(sA + (mt * 16 + m_lo) * 64, kc, khalf);
        v16h Bf = load_b_frag(wpi + (((kc * 4 + w) * 32 + lane) << 4));
        acc = __builtin_amdgcn_wmma_f32_16x16x32_f16(false, A, false, Bf,
                                                     (short)0, acc, false, false);
      }
#pragma unroll
      for (int rr = 0; rr < 8; ++rr) {
        float v = acc[rr] + bb;
        f0[mt][rr] = v;
        float lv = v > 0.f ? v : 0.01f * v;
        sB[(mt * 16 + rr + khalf * 8) * 64 + ch] = (f16)lv;
      }
    }
  }
  __syncthreads();

  // GEMM2: h1 = leaky(f0) @ w_p1 + b_p1 ; write leaky(h1) -> sA
  {
    float bb = bp1[ch];
#pragma unroll
    for (int mt = 0; mt < 4; ++mt) {
      v8f acc = {};
#pragma unroll
      for (int kc = 0; kc < 2; ++kc) {
        v16h A = load_a_frag(sB + (mt * 16 + m_lo) * 64, kc, khalf);
        v16h Bf = load_b_frag(wp1 + (((kc * 4 + w) * 32 + lane) << 4));
        acc = __builtin_amdgcn_wmma_f32_16x16x32_f16(false, A, false, Bf,
                                                     (short)0, acc, false, false);
      }
#pragma unroll
      for (int rr = 0; rr < 8; ++rr) {
        float v = acc[rr] + bb;
        float lv = v > 0.f ? v : 0.01f * v;
        sA[(mt * 16 + rr + khalf * 8) * 64 + ch] = (f16)lv;
      }
    }
  }
  __syncthreads();

  // GEMM3: h2 = leaky(h1) @ w_p2 + b_p2 ; pts = (h2 + f0)/sqrt2 -> sP
  {
    float bb = bp2[ch];
#pragma unroll
    for (int mt = 0; mt < 4; ++mt) {
      v8f acc = {};
#pragma unroll
      for (int kc = 0; kc < 2; ++kc) {
        v16h A = load_a_frag(sA + (mt * 16 + m_lo) * 64, kc, khalf);
        v16h Bf = load_b_frag(wp2 + (((kc * 4 + w) * 32 + lane) << 4));
        acc = __builtin_amdgcn_wmma_f32_16x16x32_f16(false, A, false, Bf,
                                                     (short)0, acc, false, false);
      }
#pragma unroll
      for (int rr = 0; rr < 8; ++rr) {
        float v = (acc[rr] + bb + f0[mt][rr]) * INV_SQRT2;
        sP[(mt * 16 + rr + khalf * 8) * 64 + ch] = v;
      }
    }
  }
  __syncthreads();

  // Devoxelize: 2 threads per point, 32 channels each
  int pt = tid >> 1;
  int cb = (tid & 1) * 32;
  const float* pp = points + ((long long)b * N_ + p0 + pt) * 3;
  int ix, iy, iz; float fx, fy, fz;
  corners3(pp[0], pp[1], pp[2], ix, iy, iz, fx, fy, fz);
  float dv[32];
#pragma unroll
  for (int c = 0; c < 32; ++c) dv[c] = 0.f;
  long long gb = (long long)b * ((long long)V_ * 64);
#pragma unroll
  for (int k = 0; k < 8; ++k) {
    int ox = (k >> 2) & 1, oy = (k >> 1) & 1, oz = k & 1;
    float wk = (ox ? fx : 1.f - fx) * (oy ? fy : 1.f - fy) * (oz ? fz : 1.f - fz);
    long long v = ((long long)(ix + ox) * R_ + (iy + oy)) * R_ + (iz + oz);
    const float* gp = out_grid + gb + v * 64 + cb;
#pragma unroll
    for (int c = 0; c < 32; ++c) dv[c] += wk * gp[c];
  }
  long long ob = ((long long)b * N_ + p0 + pt) * 64 + cb;
#pragma unroll
  for (int c = 0; c < 32; ++c)
    out[ob + c] = (sP[pt * 64 + cb + c] + dv[c]) * INV_SQRT2;
}

// ---------------------------------------------------------------------------
// Host-side orchestration
// ---------------------------------------------------------------------------
extern "C" void kernel_launch(void* const* d_in, const int* in_sizes, int n_in,
                              void* d_out, int out_size, void* d_ws, size_t ws_size,
                              hipStream_t stream) {
  const float* points   = (const float*)d_in[0];
  const float* features = (const float*)d_in[1];
  const float* w_pi = (const float*)d_in[2];
  const float* b_pi = (const float*)d_in[3];
  const float* w_p1 = (const float*)d_in[4];
  const float* b_p1 = (const float*)d_in[5];
  const float* w_p2 = (const float*)d_in[6];
  const float* b_p2 = (const float*)d_in[7];
  const float* g1   = (const float*)d_in[8];
  const float* be1  = (const float*)d_in[9];
  const float* w_c1 = (const float*)d_in[10];
  const float* b_c1 = (const float*)d_in[11];
  const float* g2   = (const float*)d_in[12];
  const float* be2  = (const float*)d_in[13];
  const float* w_c2 = (const float*)d_in[14];
  const float* b_c2 = (const float*)d_in[15];
  float* out = (float*)d_out;

  const long long BVC = (long long)B_ * V_ * 64;   // 16,777,216
  char* ws = (char*)d_ws;
  size_t off = 0;
  auto carve = [&](size_t bytes) -> char* {
    char* p = ws + off;
    off += (bytes + 255) & ~(size_t)255;
    return p;
  };
  float* grid  = (float*)carve(BVC * 4);            // input_grid (accum -> normalized)
  float* convo = (float*)carve(BVC * 4);            // conv1 out, then out_grid
  f16*   act   = (f16*)  carve(BVC * 2);            // f16 activations (reused)
  float* cnt   = (float*)carve((long long)B_ * V_ * 4);
  float* sacc1 = (float*)carve(512 * 4);
  float* sacc2 = (float*)carve(512 * 4);
  float* sfin1 = (float*)carve(512 * 4);
  float* sfin2 = (float*)carve(512 * 4);
  f16* wpc1 = (f16*)carve(110592 * 2);
  f16* wpc2 = (f16*)carve(110592 * 2);
  f16* wppi = (f16*)carve(4096 * 2);
  f16* wpp1 = (f16*)carve(4096 * 2);
  f16* wpp2 = (f16*)carve(4096 * 2);

  // zero accumulators
  pv_zero_kernel<<<4096, 256, 0, stream>>>(grid, BVC);
  pv_zero_kernel<<<512, 256, 0, stream>>>(cnt, (long long)B_ * V_);
  pv_zero_kernel<<<1, 256, 0, stream>>>(sacc1, 512);
  pv_zero_kernel<<<1, 256, 0, stream>>>(sacc2, 512);

  // pack weights into WMMA-B fragment layout (f16)
  pv_pack_mat64_kernel<<<16, 256, 0, stream>>>(w_pi, wppi);
  pv_pack_mat64_kernel<<<16, 256, 0, stream>>>(w_p1, wpp1);
  pv_pack_mat64_kernel<<<16, 256, 0, stream>>>(w_p2, wpp2);
  pv_pack_convw_kernel<<<432, 256, 0, stream>>>(w_c1, wpc1);
  pv_pack_convw_kernel<<<432, 256, 0, stream>>>(w_c2, wpc2);

  // voxelize
  pv_scatter_kernel<<<8192, 256, 0, stream>>>(points, features, grid, cnt);

  // GN1 (normalizes grid in-place while reducing) -> act
  pv_gn_stats_kernel<<<4096, 256, 0, stream>>>(grid, cnt, grid, sacc1);
  pv_gn_finalize_kernel<<<1, 256, 0, stream>>>(sacc1, sfin1);
  pv_act_kernel<<<65536, 256, 0, stream>>>(grid, sfin1, g1, be1, act);

  // conv1
  pv_conv3d_wmma_kernel<<<8192, 128, 0, stream>>>(act, wpc1, b_c1, nullptr, convo);

  // GN2 -> act
  pv_gn_stats_kernel<<<4096, 256, 0, stream>>>(convo, nullptr, nullptr, sacc2);
  pv_gn_finalize_kernel<<<1, 256, 0, stream>>>(sacc2, sfin2);
  pv_act_kernel<<<65536, 256, 0, stream>>>(convo, sfin2, g2, be2, act);

  // conv2 with skip: out_grid = (conv2 + input_grid)/sqrt2  (overwrites convo)
  pv_conv3d_wmma_kernel<<<8192, 128, 0, stream>>>(act, wpc2, b_c2, grid, convo);

  // fused point MLP (3 WMMA GEMMs) + devoxelize + final combine
  pv_point_devox_kernel<<<4096, 128, 0, stream>>>(points, features, wppi, wpp1, wpp2,
                                                  b_pi, b_p1, b_p2, convo, out);
}
